// AttentionLayer_53592601920071
// MI455X (gfx1250) — matrix operand order repaired and checked by reading, not compiled, not run on hardware
//
#include <hip/hip_runtime.h>

// ---------------------------------------------------------------------------
// AttentionLayer for MI455X (gfx1250, wave32, WMMA + TDM).
// Pipeline:
//   K1 prep_weights : Wqkv,Wout -> bf16, K-transposed (B-operand layout)
//   K2 qkv_gemm     : qkv = x @ Wqkv via v_wmma_f32_16x16x32_bf16; B tiles
//                     DMA'd into LDS by the Tensor Data Mover (double-buffered)
//   K3 qrpe_kernel  : qrpe[b,h,v,j] = q . rpe_emb[j]  (10 hops only)
//   K4 attn_kernel  : S = qk^T (WMMA) + qrpe gather; exact softmax (wave32
//                     shuffles); P = alpha*softmax + outer; out = P @ v (WMMA)
//   K5 out_gemm     : out = hob @ Wout + bout (WMMA + TDM staging)
// ---------------------------------------------------------------------------

#define DEV __device__ __forceinline__

typedef __attribute__((ext_vector_type(16))) __bf16 v16bf;
typedef __attribute__((ext_vector_type(8)))  __bf16 v8bf;
typedef __attribute__((ext_vector_type(8)))  float  v8f;
typedef __attribute__((ext_vector_type(4)))  float  v4f;
typedef __attribute__((ext_vector_type(4)))  unsigned u32x4;
typedef __attribute__((ext_vector_type(8)))  int    i32x8;
typedef __attribute__((ext_vector_type(4)))  int    i32x4;

constexpr int BB = 16, VV = 512, DD = 512, HH = 8, HD = 64, NH = 10;
constexpr int BV = BB * VV;   // 8192 rows
constexpr int D3 = 3 * DD;    // 1536

#if defined(__gfx1250__) && __has_builtin(__builtin_amdgcn_tensor_load_to_lds)
#define HAVE_TDM 1
#endif

// D = A(16x32 bf16) * B(32x16 bf16) + C(16x16 f32)
DEV v8f wmma_bf16(v16bf a, v16bf b, v8f c) {
  return __builtin_amdgcn_wmma_f32_16x16x32_bf16(
      /*neg_a=*/false, a, /*neg_b=*/false, b,
      /*c_mod=*/(short)0, c, /*reuse_a=*/false, /*reuse_b=*/false);
}

// A fragment (16x32 bf16, row-major source): lane L(0..15)=row M=L,
// elems 0..15 -> K=kh16+0..15, kh16=16*(lane>>4).
DEV v16bf afrag_bf(const __bf16* base, int ld, int m0, int k0, int lane) {
  const __bf16* r = base + (size_t)(m0 + (lane & 15)) * ld + k0 + ((lane >> 4) * 16);
  v8bf lo = *(const v8bf*)r;
  v8bf hi = *(const v8bf*)(r + 8);
  return __builtin_shufflevector(lo, hi, 0,1,2,3,4,5,6,7,8,9,10,11,12,13,14,15);
}

// A fragment from fp32 source with on-the-fly bf16 conversion.
DEV v16bf afrag_f32(const float* base, int ld, int m0, int k0, int lane) {
  const float* r = base + (size_t)(m0 + (lane & 15)) * ld + k0 + ((lane >> 4) * 16);
  v4f f0 = *(const v4f*)r;
  v4f f1 = *(const v4f*)(r + 4);
  v4f f2 = *(const v4f*)(r + 8);
  v4f f3 = *(const v4f*)(r + 12);
  v16bf a;
  a[0]=(__bf16)f0[0]; a[1]=(__bf16)f0[1]; a[2]=(__bf16)f0[2]; a[3]=(__bf16)f0[3];
  a[4]=(__bf16)f1[0]; a[5]=(__bf16)f1[1]; a[6]=(__bf16)f1[2]; a[7]=(__bf16)f1[3];
  a[8]=(__bf16)f2[0]; a[9]=(__bf16)f2[1]; a[10]=(__bf16)f2[2]; a[11]=(__bf16)f2[3];
  a[12]=(__bf16)f3[0]; a[13]=(__bf16)f3[1]; a[14]=(__bf16)f3[2]; a[15]=(__bf16)f3[3];
  return a;
}

// B fragment (32x16 KxN, K-transposed storage): rowp -> Bt row n (K contiguous),
// element e -> K = off + e, off includes 16*(lane>>4) lane-half split.
DEV v16bf bfrag(const __bf16* rowp, int off) {
  v8bf lo = *(const v8bf*)(rowp + off);
  v8bf hi = *(const v8bf*)(rowp + off + 8);
  return __builtin_shufflevector(lo, hi, 0,1,2,3,4,5,6,7,8,9,10,11,12,13,14,15);
}

#ifdef HAVE_TDM
// TDM: DMA a [tile_y rows x tile_x bf16] 2D tile (row stride = stride elems)
// from global into LDS (packed [tile_y][tile_x]).  D# per CDNA5 ISA ch.8.
DEV void tdm_load_bf16_tile(const __bf16* g, void* lds, int tile_x, int tile_y,
                            int tensor_x, int tensor_y, int stride) {
  unsigned long long ga = (unsigned long long)(size_t)g;
  unsigned lo = (unsigned)(size_t)lds;  // generic addr low 32 bits == LDS offset
  u32x4 g0;
  g0[0] = 1u;                                              // count=1, user D#
  g0[1] = lo;                                              // lds_addr
  g0[2] = (unsigned)ga;                                    // global_addr[31:0]
  g0[3] = (unsigned)((ga >> 32) & 0x01ffffffu) | (2u << 30); // addr[56:32]|type=2
  i32x8 g1;
  g1[0] = 1 << 16;                                         // data_size = 2 bytes
  g1[1] = (int)(((unsigned)tensor_x & 0xffffu) << 16);     // tensor_dim0[15:0]
  g1[2] = (int)((((unsigned)tensor_x >> 16) & 0xffffu) |
                (((unsigned)tensor_y & 0xffffu) << 16));   // dim0[31:16]|dim1[15:0]
  g1[3] = (int)((((unsigned)tensor_y >> 16) & 0xffffu) |
                ((unsigned)tile_x << 16));                 // dim1[31:16]|tile_dim0
  g1[4] = tile_y & 0xffff;                                 // tile_dim1, tile_dim2=0
  g1[5] = stride;                                          // tensor_dim0_stride
  g1[6] = 0;
  g1[7] = 0;
  i32x4 gz = {0, 0, 0, 0};
#if defined(__clang_major__) && __clang_major__ >= 23
  i32x8 gz8 = {0, 0, 0, 0, 0, 0, 0, 0};
  __builtin_amdgcn_tensor_load_to_lds(g0, g1, gz, gz, gz8, 0);
#else
  __builtin_amdgcn_tensor_load_to_lds(g0, g1, gz, gz, 0);
#endif
}
#endif

// ---------------------------------------------------------------------------
// K1: weights -> bf16, K-transposed.  WqkvT[n][k] = Wqkv[k][n], WoutT[n][k].
__global__ void prep_weights(const float* __restrict__ Wqkv,
                             const float* __restrict__ Wout,
                             __bf16* __restrict__ WqkvT,
                             __bf16* __restrict__ WoutT) {
  int t = blockIdx.x * blockDim.x + threadIdx.x;
  if (t < D3 * DD) {
    int n = t / DD, k = t % DD;
    WqkvT[(size_t)n * DD + k] = (__bf16)Wqkv[(size_t)k * D3 + n];
  } else {
    int u = t - D3 * DD;
    if (u < DD * DD) {
      int n = u / DD, k = u % DD;
      WoutT[(size_t)n * DD + k] = (__bf16)Wout[(size_t)k * DD + n];
    }
  }
}

// ---------------------------------------------------------------------------
// K2: qkv = x @ Wqkv. Block tile 128(M) x 64(N); 8 waves, each wave 16x64.
// B tiles staged in LDS via TDM (double-buffered); A cvt fp32->bf16 in regs.
__global__ __launch_bounds__(256)
void qkv_gemm(const float* __restrict__ x, const __bf16* __restrict__ WqkvT,
              __bf16* __restrict__ qb, __bf16* __restrict__ kb,
              __bf16* __restrict__ vT) {
  __shared__ __bf16 BtL[2][64 * 32];
  const int lane = threadIdx.x & 31;
  const int wi   = threadIdx.x >> 5;
  const int m0   = blockIdx.x * 128 + wi * 16;
  const int c0   = blockIdx.y * 64;
  const int which = c0 >> 9;          // 0:q 1:k 2:v (constant per block)
  const int h     = (c0 & 511) >> 6;  // head (constant per block)
  const int kh16 = (lane >> 4) * 16;
  const int nlb  = lane & 15;

  v8f acc[4] = {};
#ifdef HAVE_TDM
  if (wi == 0) {
    tdm_load_bf16_tile(WqkvT + (size_t)c0 * DD, BtL[0], 32, 64, DD, 64, DD);
    __builtin_amdgcn_s_wait_tensorcnt(0);
  }
  __syncthreads();
  for (int ks = 0; ks < 16; ks++) {
    int k0 = ks * 32;
    const __bf16* Bcur = BtL[ks & 1];
    if (wi == 0 && ks + 1 < 16)   // DMA next tile while computing this one
      tdm_load_bf16_tile(WqkvT + (size_t)c0 * DD + k0 + 32, BtL[(ks + 1) & 1],
                         32, 64, DD, 64, DD);
    v16bf a = afrag_f32(x, DD, m0, k0, lane);
    v16bf bfr[4];
    for (int ns = 0; ns < 4; ns++)
      bfr[ns] = bfrag(&Bcur[(ns * 16 + nlb) * 32], kh16);
    for (int ns = 0; ns < 4; ns++)
      acc[ns] = wmma_bf16(a, bfr[ns], acc[ns]);
    if (wi == 0) __builtin_amdgcn_s_wait_tensorcnt(0);
    __syncthreads();
  }
#else
  for (int ks = 0; ks < 16; ks++) {
    int k0 = ks * 32;
    __bf16* Bcur = BtL[ks & 1];
    __syncthreads();
    {
      int row = threadIdx.x >> 2, c8 = (threadIdx.x & 3) * 8;
      *(v8bf*)&Bcur[row * 32 + c8] =
          *(const v8bf*)&WqkvT[(size_t)(c0 + row) * DD + k0 + c8];
    }
    __syncthreads();
    v16bf a = afrag_f32(x, DD, m0, k0, lane);
    v16bf bfr[4];
    for (int ns = 0; ns < 4; ns++)
      bfr[ns] = bfrag(&Bcur[(ns * 16 + nlb) * 32], kh16);
    for (int ns = 0; ns < 4; ns++)
      acc[ns] = wmma_bf16(a, bfr[ns], acc[ns]);
  }
#endif

  const int mh8 = (lane >> 4) * 8;
  const int grow = m0 + mh8;          // global row of acc element r=0
  const int bb = grow >> 9;           // batch
  const int vvb = grow & 511;         // vertex base (8 consecutive)
  for (int ns = 0; ns < 4; ns++) {
    int nl = ns * 16 + nlb;           // 0..63 == hd (c0 is 64-aligned)
    if (which == 2) {                 // v -> K-transposed layout, packed store
      v8bf pv;
      for (int r = 0; r < 8; r++) pv[r] = (__bf16)acc[ns][r];
      *(v8bf*)&vT[(((size_t)bb * HH + h) * HD + nl) * VV + vvb] = pv;
    } else {
      __bf16* dst = (which == 0) ? qb : kb;
      for (int r = 0; r < 8; r++)
        dst[(((size_t)bb * HH + h) * VV + (vvb + r)) * HD + nl] =
            (__bf16)acc[ns][r];
    }
  }
}

// ---------------------------------------------------------------------------
// K3: qrpe[row][j] = sum_d q[row][d] * rpe[j][d], row = (b,h,v), j padded to 16.
__global__ void qrpe_kernel(const __bf16* __restrict__ qb,
                            const float* __restrict__ rpe,
                            float* __restrict__ qrpe) {
  int g = blockIdx.x * blockDim.x + threadIdx.x;
  int row = g >> 4, j = g & 15;
  if (row >= BB * HH * VV) return;
  float s = 0.f;
  if (j < NH) {
    const __bf16* q = qb + (size_t)row * HD;
    for (int d = 0; d < HD; d++) s += (float)q[d] * rpe[j * HD + d];
  }
  qrpe[(size_t)row * 16 + j] = s;
}

// ---------------------------------------------------------------------------
// K4: attention. Block = 4 waves x 16 query rows; each wave independent.
// LDS per wave: 16x512 fp32 scores (reused in-place as 16x512 bf16 P).
__global__ __launch_bounds__(128)
void attn_kernel(const __bf16* __restrict__ qb, const __bf16* __restrict__ kb,
                 const __bf16* __restrict__ vT, const float* __restrict__ qrpe,
                 const int* __restrict__ hop, const float* __restrict__ outer,
                 const float* __restrict__ alpha_p, __bf16* __restrict__ hob) {
  extern __shared__ char smem[];
  const int lane = threadIdx.x & 31;
  const int wi   = threadIdx.x >> 5;   // 0..3
  const int rb   = blockIdx.x;         // 0..7
  const int bh   = blockIdx.y;         // 0..127
  const int b    = bh >> 3, h = bh & 7;
  const int v0   = rb * 64 + wi * 16;

  float* S  = (float*)smem + (size_t)wi * 16 * 512;
  float* qr = (float*)(smem + 4 * 16 * 512 * 4) + wi * 16 * 16;

  // stage this wave's qrpe tile [16 rows][16 slots]
  for (int i = 0; i < 8; i++) {
    int idx = i * 32 + lane;
    qr[idx] = qrpe[((size_t)bh * VV + v0 + (idx >> 4)) * 16 + (idx & 15)];
  }

  const __bf16* qbh = qb + (size_t)bh * VV * HD;
  const __bf16* kbh = kb + (size_t)bh * VV * HD;
  const __bf16* vbh = vT + (size_t)bh * HD * VV;

  v16bf aQ0 = afrag_bf(qbh, HD, v0, 0, lane);
  v16bf aQ1 = afrag_bf(qbh, HD, v0, 32, lane);

  const int kh16 = (lane >> 4) * 16;
  const int nlb  = lane & 15;
  const int mh8  = (lane >> 4) * 8;

  // S = q @ k^T, streamed to LDS in 64-column tiles (loads batched ahead of
  // the WMMA burst so the loadcnt drains progressively).
  for (int wt = 0; wt < 8; wt++) {
    int w0 = wt * 64;
    v8f acc[4] = {};
    v16bf bfr[8];
    for (int ns = 0; ns < 4; ns++) {
      const __bf16* rowp = kbh + (size_t)(w0 + ns * 16 + nlb) * HD;
      bfr[ns]     = bfrag(rowp, kh16);
      bfr[4 + ns] = bfrag(rowp, 32 + kh16);
    }
    for (int ns = 0; ns < 4; ns++) acc[ns] = wmma_bf16(aQ0, bfr[ns], acc[ns]);
    for (int ns = 0; ns < 4; ns++) acc[ns] = wmma_bf16(aQ1, bfr[4 + ns], acc[ns]);
    for (int ns = 0; ns < 4; ns++)
      for (int r = 0; r < 8; r++)
        S[(r + mh8) * 512 + w0 + ns * 16 + nlb] = acc[ns][r];
  }

  const float scale = 0.125f;    // HD^-0.5
  const float al = alpha_p[0];
  __bf16* P = (__bf16*)S;        // in-place fp32 -> bf16 (rows ascend => safe)

  for (int rr = 0; rr < 16; rr++) {
    int v = v0 + rr;
    float vals[16];
    float m = -3.0e38f;
    for (int i = 0; i < 16; i++) {
      int w = i * 32 + lane;
      float s = S[rr * 512 + w] + qr[rr * 16 + hop[(size_t)v * VV + w]];
      s *= scale;
      vals[i] = s;
      m = fmaxf(m, s);
    }
    for (int o = 16; o > 0; o >>= 1) m = fmaxf(m, __shfl_xor(m, o, 32));
    float sum = 0.f;
    for (int i = 0; i < 16; i++) { vals[i] = __expf(vals[i] - m); sum += vals[i]; }
    for (int o = 16; o > 0; o >>= 1) sum += __shfl_xor(sum, o, 32);
    float inv = 1.0f / sum;
    const float* orow = outer + ((size_t)h * VV + v) * VV;
    for (int i = 0; i < 16; i++) {
      int w = i * 32 + lane;
      P[rr * 512 + w] = (__bf16)(al * vals[i] * inv + orow[w]);
    }
  }

  // out = P @ v   (A from LDS bf16, B from global vT), 2x K-unrolled
  v8f oacc[4] = {};
  for (int k0 = 0; k0 < VV; k0 += 64) {
    v16bf a0 = afrag_bf(P, 512, 0, k0, lane);
    v16bf a1 = afrag_bf(P, 512, 0, k0 + 32, lane);
    v16bf bfr[8];
    for (int ns = 0; ns < 4; ns++) {
      const __bf16* rowp = vbh + (size_t)(ns * 16 + nlb) * VV;
      bfr[ns]     = bfrag(rowp, k0 + kh16);
      bfr[4 + ns] = bfrag(rowp, k0 + 32 + kh16);
    }
    for (int ns = 0; ns < 4; ns++) oacc[ns] = wmma_bf16(a0, bfr[ns], oacc[ns]);
    for (int ns = 0; ns < 4; ns++) oacc[ns] = wmma_bf16(a1, bfr[4 + ns], oacc[ns]);
  }
  for (int ns = 0; ns < 4; ns++) {
    int col = h * HD + ns * 16 + nlb;
    for (int r = 0; r < 8; r++) {
      int g = b * VV + v0 + mh8 + r;
      hob[(size_t)g * DD + col] = (__bf16)oacc[ns][r];
    }
  }
}

// ---------------------------------------------------------------------------
// K5: out = hob @ Wout + bout (fp32 result).  Same TDM staging as K2.
__global__ __launch_bounds__(256)
void out_gemm(const __bf16* __restrict__ hob, const __bf16* __restrict__ WoutT,
              const float* __restrict__ bout, float* __restrict__ out) {
  __shared__ __bf16 BtL[2][64 * 32];
  const int lane = threadIdx.x & 31;
  const int wi   = threadIdx.x >> 5;
  const int m0   = blockIdx.x * 128 + wi * 16;
  const int c0   = blockIdx.y * 64;
  const int kh16 = (lane >> 4) * 16;
  const int nlb  = lane & 15;

  v8f acc[4] = {};
#ifdef HAVE_TDM
  if (wi == 0) {
    tdm_load_bf16_tile(WoutT + (size_t)c0 * DD, BtL[0], 32, 64, DD, 64, DD);
    __builtin_amdgcn_s_wait_tensorcnt(0);
  }
  __syncthreads();
  for (int ks = 0; ks < 16; ks++) {
    int k0 = ks * 32;
    const __bf16* Bcur = BtL[ks & 1];
    if (wi == 0 && ks + 1 < 16)
      tdm_load_bf16_tile(WoutT + (size_t)c0 * DD + k0 + 32, BtL[(ks + 1) & 1],
                         32, 64, DD, 64, DD);
    v16bf a = afrag_bf(hob, DD, m0, k0, lane);
    v16bf bfr[4];
    for (int ns = 0; ns < 4; ns++)
      bfr[ns] = bfrag(&Bcur[(ns * 16 + nlb) * 32], kh16);
    for (int ns = 0; ns < 4; ns++)
      acc[ns] = wmma_bf16(a, bfr[ns], acc[ns]);
    if (wi == 0) __builtin_amdgcn_s_wait_tensorcnt(0);
    __syncthreads();
  }
#else
  for (int ks = 0; ks < 16; ks++) {
    int k0 = ks * 32;
    __bf16* Bcur = BtL[ks & 1];
    __syncthreads();
    {
      int row = threadIdx.x >> 2, c8 = (threadIdx.x & 3) * 8;
      *(v8bf*)&Bcur[row * 32 + c8] =
          *(const v8bf*)&WoutT[(size_t)(c0 + row) * DD + k0 + c8];
    }
    __syncthreads();
    v16bf a = afrag_bf(hob, DD, m0, k0, lane);
    v16bf bfr[4];
    for (int ns = 0; ns < 4; ns++)
      bfr[ns] = bfrag(&Bcur[(ns * 16 + nlb) * 32], kh16);
    for (int ns = 0; ns < 4; ns++)
      acc[ns] = wmma_bf16(a, bfr[ns], acc[ns]);
  }
#endif

  const int mh8 = (lane >> 4) * 8;
  for (int ns = 0; ns < 4; ns++) {
    int c = c0 + ns * 16 + nlb;
    float bv = bout[c];
    for (int r = 0; r < 8; r++)
      out[(size_t)(m0 + mh8 + r) * DD + c] = acc[ns][r] + bv;
  }
}

// ---------------------------------------------------------------------------
extern "C" void kernel_launch(void* const* d_in, const int* in_sizes, int n_in,
                              void* d_out, int out_size, void* d_ws,
                              size_t ws_size, hipStream_t stream) {
  const float* x     = (const float*)d_in[0];
  const float* Wqkv  = (const float*)d_in[1];
  const float* Wout  = (const float*)d_in[2];
  const float* bout  = (const float*)d_in[3];
  const float* rpe   = (const float*)d_in[4];
  const float* outer = (const float*)d_in[5];
  const float* alpha = (const float*)d_in[6];
  const int*   hop   = (const int*)d_in[7];
  float* out = (float*)d_out;

  char* ws = (char*)d_ws;
  size_t o = 0;
  __bf16* WqkvT = (__bf16*)(ws + o); o += (size_t)D3 * DD * 2;
  __bf16* WoutT = (__bf16*)(ws + o); o += (size_t)DD * DD * 2;
  __bf16* qb    = (__bf16*)(ws + o); o += (size_t)BB * HH * VV * HD * 2;
  __bf16* kb    = (__bf16*)(ws + o); o += (size_t)BB * HH * VV * HD * 2;
  __bf16* vT    = (__bf16*)(ws + o); o += (size_t)BB * HH * VV * HD * 2;
  float*  qrp   = (float*)(ws + o);  o += (size_t)BB * HH * VV * 16 * 4;
  __bf16* hob   = (__bf16*)(ws + o); o += (size_t)BV * DD * 2;

  prep_weights<<<4096, 256, 0, stream>>>(Wqkv, Wout, WqkvT, WoutT);
  qkv_gemm<<<dim3(BV / 128, D3 / 64), 256, 0, stream>>>(x, WqkvT, qb, kb, vT);
  qrpe_kernel<<<(BB * HH * VV * 16) / 256, 256, 0, stream>>>(qb, rpe, qrp);
  attn_kernel<<<dim3(8, BB * HH), 128, 4 * 16 * 512 * 4 + 4 * 16 * 16 * 4,
                stream>>>(qb, kb, vT, qrp, hop, outer, alpha, hob);
  out_gemm<<<dim3(BV / 128, DD / 64), 256, 0, stream>>>(hob, WoutT, bout, out);
}
